// SubGraphEncoder_34857954575031
// MI455X (gfx1250) — matrix operand order repaired and checked by reading, not compiled, not run on hardware
//
#include <hip/hip_runtime.h>

#define B_   32
#define M_   1024
#define DIN  128
#define DOUT 128
#define EDIM 64

typedef __attribute__((ext_vector_type(16))) _Float16 v16h;
typedef __attribute__((ext_vector_type(8)))  float    v8f;
typedef unsigned short u16;
typedef unsigned int   u32;

union Frag  { v16h v; u32 u[8]; _Float16 h[16]; };
union Acc   { v8f  v; float f[8]; };
union Pack8 { int4 q; _Float16 h[8]; };

__device__ __forceinline__ u16 f2h_bits(float x) {
    union { _Float16 h; u16 s; } cv; cv.h = (_Float16)x; return cv.s;
}

// A-matrix fragment (16x32 f16), row-major source, K-pairs contiguous.
// lanes 0-15: row=lane, V0..V3 -> K=0..7, V4..V7 -> K=16..23
// lanes16-31: row=lane-16,  V0..V3 -> K=8..15, V4..V7 -> K=24..31
__device__ __forceinline__ v16h load_A(const u16* __restrict__ src, int ld,
                                       int row0, int k0, int lane) {
    const int hs = lane >> 4, r = lane & 15;
    const u16* base = src + (size_t)(row0 + r) * ld + k0 + hs * 8;
    Frag f;
#pragma unroll
    for (int v = 0; v < 4; ++v) f.u[v]     = *(const u32*)(base + 2 * v);
#pragma unroll
    for (int v = 0; v < 4; ++v) f.u[4 + v] = *(const u32*)(base + 16 + 2 * v);
    return f.v;
}

// B-matrix fragment (32x16 f16) read from a K-contiguous ("transposed") source:
// element B[k][n] lives at src[n*ld + k].
// lanes 0-15: n=lane, K=0..7,16..23 ; lanes 16-31: n=lane-16, K=8..15,24..31
__device__ __forceinline__ v16h load_B(const u16* __restrict__ src, int ld,
                                       int n0, int k0, int lane) {
    const int hs = lane >> 4, n = lane & 15;
    const u16* base = src + (size_t)(n0 + n) * ld + k0 + hs * 8;
    Frag f;
#pragma unroll
    for (int v = 0; v < 8; ++v) f.u[v] = *(const u32*)(base + 2 * v + (v >> 2) * 8);
    return f.v;
}

// ---------------- elementwise prep kernels ----------------
__global__ void zero_f_kernel(float* p, int n) {
    int i = blockIdx.x * blockDim.x + threadIdx.x;
    if (i < n) p[i] = 0.0f;
}
__global__ void f32_to_f16_kernel(const float* __restrict__ src, u16* __restrict__ dst, int n) {
    int i = blockIdx.x * blockDim.x + threadIdx.x;
    if (i < n) dst[i] = f2h_bits(src[i]);
}
__global__ void rsqrt_kernel(const float* __restrict__ d, float* __restrict__ dinv, int n) {
    int i = blockIdx.x * blockDim.x + threadIdx.x;
    if (i < n) dinv[i] = rsqrtf(d[i]);
}
// XsT[b][f][n] = dinv[b][n] * X[b][n][f]   (f16, node index fastest -> coalesced writes)
__global__ void make_xst_kernel(const float* __restrict__ X, const float* __restrict__ dinv,
                                u16* __restrict__ XsT) {
    int i = blockIdx.x * blockDim.x + threadIdx.x;   // i = (b*DIN + f)*M + n
    int n = i & (M_ - 1);
    int f = (i >> 10) & (DIN - 1);
    int b = i >> 17;
    float x = X[((size_t)b * M_ + n) * DIN + f];
    XsT[i] = f2h_bits(dinv[(size_t)b * M_ + n] * x);
}

// ---------------- adjacency: adj = relu(E E^T) + I, row sums d ----------------
__global__ __launch_bounds__(256)
void adj_kernel(const u16* __restrict__ Ef,   // [B][M][EDIM] f16
                u16* __restrict__ adj,        // [B][M][M]    f16
                float* __restrict__ d)        // [B][M]       f32 (pre-zeroed)
{
    const int cg = blockIdx.x;        // 0..7   -> 128 columns
    const int st = blockIdx.y;        // 0..15  -> 64 rows
    const int b  = blockIdx.z;
    const int wave = threadIdx.x >> 5;
    const int lane = threadIdx.x & 31;
    const int rw = wave & 3, cw = wave >> 2;
    const int row0 = st * 64 + rw * 16;
    const int col0 = cg * 128 + cw * 64;
    const int hs = lane >> 4, nl = lane & 15;

    const u16* Eb = Ef + (size_t)b * M_ * EDIM;

    Acc acc[4];
#pragma unroll
    for (int t = 0; t < 4; ++t) { v8f z = {0,0,0,0,0,0,0,0}; acc[t].v = z; }

#pragma unroll
    for (int k0 = 0; k0 < EDIM; k0 += 32) {
        v16h a = load_A(Eb, EDIM, row0, k0, lane);
#pragma unroll
        for (int t = 0; t < 4; ++t) {
            v16h bb = load_B(Eb, EDIM, col0 + t * 16, k0, lane); // B[k][n]=E[n][k]
            acc[t].v = __builtin_amdgcn_wmma_f32_16x16x32_f16(
                false, a, false, bb, (short)0, acc[t].v, false, false);
        }
    }

    const int mbase = row0 + hs * 8;   // 8 consecutive M values per lane (across vgprs)
    float rsum[8];
#pragma unroll
    for (int v = 0; v < 8; ++v) rsum[v] = 0.0f;

    u16* adjb = adj + (size_t)b * M_ * M_;
#pragma unroll
    for (int t = 0; t < 4; ++t) {
        const int n = col0 + t * 16 + nl;
        Pack8 p;
#pragma unroll
        for (int v = 0; v < 8; ++v) {
            float c = acc[t].f[v];
            c = c > 0.0f ? c : 0.0f;
            if (mbase + v == n) c += 1.0f;          // + identity
            rsum[v] += c;
            p.h[v] = (_Float16)c;
        }
        // adj is symmetric: store tile element (m,n) at [n][m] -> contiguous 16B
        *(int4*)(adjb + (size_t)n * M_ + mbase) = p.q;
    }
    // reduce partial row-sums across 16 lanes of each half-wave (same M set)
#pragma unroll
    for (int v = 0; v < 8; ++v) {
#pragma unroll
        for (int msk = 8; msk >= 1; msk >>= 1)
            rsum[v] += __shfl_xor(rsum[v], msk, 32);
    }
    if (nl == 0) {
        float* db = d + (size_t)b * M_;
#pragma unroll
        for (int v = 0; v < 8; ++v) atomicAdd(&db[mbase + v], rsum[v]);
    }
}

// ---------------- fused GCN layer: Y = relu( (dinv_m * (adj @ Xs)) @ W^T ) ----------------
// mode 0: write outT[b][f][m] = dinv[m] * Y[m][f]  (f16, transposed input for next layer)
// mode 1: write outF[b][m][f] = Y[m][f]            (f32, final output)
__global__ __launch_bounds__(256)
void layer_kernel(const u16* __restrict__ adj,    // [B][M][M]  f16 (symmetric)
                  const u16* __restrict__ XsT,    // [B][128][M] f16
                  const u16* __restrict__ Wf,     // [128][128] f16 row-major (B[k][j]=W[j][k])
                  const float* __restrict__ dinv, // [B][M]
                  u16* __restrict__ outT,
                  float* __restrict__ outF,
                  int mode)
{
    __shared__ u16  Hs[64 * 136];   // 64 rows x 128 cols, +8 half pad vs banks
    __shared__ float dv[64];

    const int st = blockIdx.x;      // 0..15 -> 64-row stripe
    const int b  = blockIdx.y;
    const int wave = threadIdx.x >> 5;
    const int lane = threadIdx.x & 31;
    const int rw = wave & 3, cw = wave >> 2;
    const int hs = lane >> 4, nl = lane & 15;

    if (threadIdx.x < 64)
        dv[threadIdx.x] = dinv[(size_t)b * M_ + st * 64 + threadIdx.x];
    __syncthreads();

    const u16* adjb = adj + ((size_t)b * M_ + st * 64 + rw * 16) * M_;
    const u16* Xb   = XsT + (size_t)b * DIN * M_;

    // ---- GEMM1: H = adj(64xM) @ Xs(Mx128), K = 1024 ----
    Acc acc[4];
#pragma unroll
    for (int t = 0; t < 4; ++t) { v8f z = {0,0,0,0,0,0,0,0}; acc[t].v = z; }

    for (int k0 = 0; k0 < M_; k0 += 32) {
        v16h a = load_A(adjb, M_, 0, k0, lane);
#pragma unroll
        for (int t = 0; t < 4; ++t) {
            v16h bb = load_B(Xb, M_, cw * 64 + t * 16, k0, lane); // B[k][f]=XsT[f][k]
            acc[t].v = __builtin_amdgcn_wmma_f32_16x16x32_f16(
                false, a, false, bb, (short)0, acc[t].v, false, false);
        }
    }

    // scale by dinv_m, stash f16 H in LDS
    const int rloc = rw * 16 + hs * 8;
#pragma unroll
    for (int t = 0; t < 4; ++t) {
        const int cloc = cw * 64 + t * 16 + nl;
#pragma unroll
        for (int v = 0; v < 8; ++v) {
            float c = acc[t].f[v] * dv[rloc + v];
            Hs[(rloc + v) * 136 + cloc] = f2h_bits(c);
        }
    }
    __syncthreads();

    // ---- GEMM2: Y = H(64x128) @ W^T(128x128), K = 128 ----
    Acc acc2[4];
#pragma unroll
    for (int t = 0; t < 4; ++t) { v8f z = {0,0,0,0,0,0,0,0}; acc2[t].v = z; }

#pragma unroll
    for (int k0 = 0; k0 < DIN; k0 += 32) {
        v16h a = load_A(Hs, 136, rw * 16, k0, lane);
#pragma unroll
        for (int t = 0; t < 4; ++t) {
            v16h bb = load_B(Wf, DIN, cw * 64 + t * 16, k0, lane); // B[k][j]=W[j][k]
            acc2[t].v = __builtin_amdgcn_wmma_f32_16x16x32_f16(
                false, a, false, bb, (short)0, acc2[t].v, false, false);
        }
    }

    const int mglob = st * 64 + rloc;   // 8 consecutive global rows per lane
    if (mode == 0) {
        u16* ob = outT + (size_t)b * DOUT * M_;
#pragma unroll
        for (int t = 0; t < 4; ++t) {
            const int j = cw * 64 + t * 16 + nl;
            Pack8 p;
#pragma unroll
            for (int v = 0; v < 8; ++v) {
                float c = acc2[t].f[v];
                c = c > 0.0f ? c : 0.0f;
                c *= dv[rloc + v];               // fold next layer's dinv_n
                p.h[v] = (_Float16)c;
            }
            *(int4*)(ob + (size_t)j * M_ + mglob) = p.q;  // transposed, contiguous 16B
        }
    } else {
        float* ob = outF + ((size_t)b * M_) * DOUT;
#pragma unroll
        for (int t = 0; t < 4; ++t) {
            const int j = cw * 64 + t * 16 + nl;
#pragma unroll
            for (int v = 0; v < 8; ++v) {
                float c = acc2[t].f[v];
                ob[(size_t)(mglob + v) * DOUT + j] = c > 0.0f ? c : 0.0f;
            }
        }
    }
}

extern "C" void kernel_launch(void* const* d_in, const int* in_sizes, int n_in,
                              void* d_out, int out_size, void* d_ws, size_t ws_size,
                              hipStream_t stream) {
    (void)in_sizes; (void)n_in; (void)out_size; (void)ws_size;
    const float* X  = (const float*)d_in[0];  // [B*M, 128]
    const float* E  = (const float*)d_in[1];  // [B*M, 64]
    const float* W1 = (const float*)d_in[2];  // [128,128]
    const float* W2 = (const float*)d_in[3];  // [128,128]
    float* out = (float*)d_out;

    char* ws = (char*)d_ws;
    u16*   adjf = (u16*)(ws);                                    // 64 MB
    u16*   Ef   = (u16*)(ws + (size_t)67108864);                 //  4 MB
    u16*   XsT1 = (u16*)(ws + (size_t)71303168);                 //  8 MB
    u16*   XsT2 = (u16*)(ws + (size_t)79691776);                 //  8 MB
    u16*   W1f  = (u16*)(ws + (size_t)88080384);                 // 32 KB
    u16*   W2f  = (u16*)(ws + (size_t)88113152);                 // 32 KB
    float* dsum = (float*)(ws + (size_t)88145920);               // 128 KB
    float* dinv = (float*)(ws + (size_t)88276992);               // 128 KB

    const int nE = B_ * M_ * EDIM;     // 2,097,152
    const int nW = DOUT * DIN;         // 16,384
    const int nD = B_ * M_;            // 32,768
    const int nX = B_ * DIN * M_;      // 4,194,304

    zero_f_kernel<<<(nD + 255) / 256, 256, 0, stream>>>(dsum, nD);
    f32_to_f16_kernel<<<(nE + 255) / 256, 256, 0, stream>>>(E, Ef, nE);
    f32_to_f16_kernel<<<(nW + 255) / 256, 256, 0, stream>>>(W1, W1f, nW);
    f32_to_f16_kernel<<<(nW + 255) / 256, 256, 0, stream>>>(W2, W2f, nW);

    adj_kernel<<<dim3(8, 16, B_), 256, 0, stream>>>(Ef, adjf, dsum);
    rsqrt_kernel<<<(nD + 255) / 256, 256, 0, stream>>>(dsum, dinv, nD);
    make_xst_kernel<<<nX / 256, 256, 0, stream>>>(X, dinv, XsT1);

    layer_kernel<<<dim3(16, B_), 256, 0, stream>>>(adjf, XsT1, W1f, dinv, XsT2, nullptr, 0);
    layer_kernel<<<dim3(16, B_), 256, 0, stream>>>(adjf, XsT2, W2f, dinv, nullptr, out, 1);
}
